// Fire_15461882265620
// MI455X (gfx1250) — matrix operand order repaired and checked by reading, not compiled, not run on hardware
//
#include <hip/hip_runtime.h>

typedef __attribute__((ext_vector_type(16))) _Float16 v16h;
typedef __attribute__((ext_vector_type(8)))  _Float16 v8h;
typedef __attribute__((ext_vector_type(2)))  _Float16 v2h;
typedef __attribute__((ext_vector_type(8)))  float    v8f;
typedef __attribute__((ext_vector_type(4)))  float    v4f;

#define N_   64
#define CIN  256
#define H_   56
#define W_   56
#define HP   58          // padded rows: 1 + 56 + 1
#define WPP  66          // padded cols: 1 + 56 + 1 + 8 slack (tap shifts stay in-plane)
#define SQP  64          // squeeze channels padded 48 -> 64
#define PLANEREC ((size_t)HP * WPP)   // pixel records per image

__device__ __forceinline__ v16h cat16(v8h lo, v8h hi) {
  return __builtin_shufflevector(lo, hi, 0,1,2,3,4,5,6,7,8,9,10,11,12,13,14,15);
}

__device__ __forceinline__ float quant1(float w, float scale) {
  float q = rintf(w / scale);                 // RNE, matches jnp.round
  q = fminf(fmaxf(q, -127.f), 127.f);
  return q * scale;
}

// ---------------------------------------------------------------------------
// Weight prep: per-tensor max-abs -> symmetric int8 fake-quant -> fp16, padded
// block 0: w_sq [48][256]      -> q_sq [64][256]   (rows 48..63 zero)
// block 1: w_e1 [192][48]      -> q_e1 [192][64]   (K 48..63 zero)
// block 2: w_e3 [192][48][3][3]-> q_e3 [192][9*64] (tap-major, K 48..63 zero)
// ---------------------------------------------------------------------------
__global__ __launch_bounds__(256) void prep_kernel(
    const float* __restrict__ w_sq, const float* __restrict__ w_e1,
    const float* __restrict__ w_e3,
    _Float16* __restrict__ q_sq, _Float16* __restrict__ q_e1,
    _Float16* __restrict__ q_e3)
{
  __shared__ float red[256];
  const int which = blockIdx.x;
  const int tid = threadIdx.x;
  const float* src = (which == 0) ? w_sq : (which == 1) ? w_e1 : w_e3;
  const int nsrc = (which == 0) ? 48 * 256 : (which == 1) ? 192 * 48 : 192 * 48 * 9;

  float m = 0.f;
  for (int i = tid; i < nsrc; i += 256) m = fmaxf(m, fabsf(src[i]));
  red[tid] = m;
  __syncthreads();
  for (int s = 128; s > 0; s >>= 1) {
    if (tid < s) red[tid] = fmaxf(red[tid], red[tid + s]);
    __syncthreads();
  }
  const float scale = fmaxf(red[0] / 127.f, 1e-8f);

  if (which == 0) {
    for (int i = tid; i < 64 * 256; i += 256) {
      int o = i >> 8, k = i & 255;
      float v = 0.f;
      if (o < 48) v = quant1(w_sq[o * 256 + k], scale);
      q_sq[i] = (_Float16)v;
    }
  } else if (which == 1) {
    for (int i = tid; i < 192 * 64; i += 256) {
      int o = i >> 6, k = i & 63;
      float v = 0.f;
      if (k < 48) v = quant1(w_e1[o * 48 + k], scale);
      q_e1[i] = (_Float16)v;
    }
  } else {
    for (int i = tid; i < 192 * 576; i += 256) {
      int o = i / 576, rem = i % 576, tap = rem >> 6, k = rem & 63;
      float v = 0.f;
      if (k < 48) v = quant1(w_e3[(o * 48 + k) * 9 + tap], scale);
      q_e3[i] = (_Float16)v;
    }
  }
}

// ---------------------------------------------------------------------------
// Squeeze: 1x1 conv 256->48(pad 64) + ReLU.
// Reads NCHW fp32 x (transposed into LDS), writes zero-haloed NHWC fp16
// s_pad[n][1+h][1+w][c] so expand fragments are contiguous 32B in global.
// One block per (n,h); 4 waves = 4 out-channel tiles x 64 px.
// ---------------------------------------------------------------------------
__global__ __launch_bounds__(128) void squeeze_kernel(
    const float* __restrict__ x, const float* __restrict__ b_sq,
    const _Float16* __restrict__ Wq, _Float16* __restrict__ s_pad)
{
  __shared__ __align__(16) _Float16 Bs[64][40];   // [px][K], pitch 40 -> 16B-aligned frags
  const int nh = blockIdx.x;
  const int n = nh / H_, h = nh % H_;
  const int tid = threadIdx.x;
  const int lane = tid & 31, wv = tid >> 5;
  const int lx = lane & 15, lh = lane >> 4;

  v8f acc[4] = {};
  for (int ks = 0; ks < 8; ++ks) {
    const int kc = ks * 32;
    // --- stage 32K x 64px tile (fp32 -> fp16, transposed) ---
    {
      const int kp = tid >> 3, pg = tid & 7;          // kp: K pair, pg: 8-px group
      const int k0 = kc + 2 * kp;
      const float* r0 = x + ((size_t)(n * CIN + k0) * H_ + h) * W_ + pg * 8;
      const float* r1 = r0 + (size_t)H_ * W_;
      float a0[8], a1[8];
      if (pg < 7) {                                    // px 56..63 -> zero
        *(v4f*)&a0[0] = *(const v4f*)r0; *(v4f*)&a0[4] = *(const v4f*)(r0 + 4);
        *(v4f*)&a1[0] = *(const v4f*)r1; *(v4f*)&a1[4] = *(const v4f*)(r1 + 4);
      } else {
#pragma unroll
        for (int i = 0; i < 8; ++i) { a0[i] = 0.f; a1[i] = 0.f; }
      }
#pragma unroll
      for (int i = 0; i < 8; ++i) {
        v2h p; p.x = (_Float16)a0[i]; p.y = (_Float16)a1[i];
        *(v2h*)&Bs[pg * 8 + i][2 * kp] = p;
      }
    }
    __syncthreads();
    // --- A fragment: lane = out-ch row, 16 contiguous K: {0..15}/{16..31}
    const _Float16* wrow = Wq + (size_t)(wv * 16 + lx) * 256 + kc + lh * 16;
    const v16h a = cat16(*(const v8h*)wrow, *(const v8h*)(wrow + 8));
#pragma unroll
    for (int pt = 0; pt < 4; ++pt) {
      const _Float16* bsrc = &Bs[pt * 16 + lx][lh * 16];
      const v16h b = cat16(*(const v8h*)bsrc, *(const v8h*)(bsrc + 8));
      acc[pt] = __builtin_amdgcn_wmma_f32_16x16x32_f16(
          false, a, false, b, (short)0, acc[pt], false, false);
    }
    __syncthreads();
  }
  // --- epilogue: bias + ReLU + one b128 NHWC store per tile ---
#pragma unroll
  for (int pt = 0; pt < 4; ++pt) {
    const int px = pt * 16 + lx;
    if (px >= W_) continue;
    v8h o;
#pragma unroll
    for (int r = 0; r < 8; ++r) {
      const int c = wv * 16 + lh * 8 + r;
      const float bias = (c < 48) ? b_sq[c] : 0.f;
      float v = acc[pt][r] + bias;
      v = v > 0.f ? v : 0.f;
      o[r] = (_Float16)v;
    }
    const size_t rec = ((size_t)n * HP + (h + 1)) * WPP + (1 + px);
    *(v8h*)&s_pad[rec * SQP + wv * 16 + lh * 8] = o;
  }
}

// ---------------------------------------------------------------------------
// Expand (1x1 or 3x3): implicit GEMM, K=64 padded, no LDS / no barriers.
// Waves map to PIXEL tiles (each wave loads distinct B data; tiny weight
// A-matrix is the shared operand and lives in L0/L2).
// grid = (N*H, 3 groups of 64 out-ch); block = 128 (4 waves).
// ---------------------------------------------------------------------------
template <int TAPS, int COLOFS, int CBASE>
__global__ __launch_bounds__(128) void expand_kernel(
    const _Float16* __restrict__ s_pad, const float* __restrict__ bias,
    const _Float16* __restrict__ Wq, float* __restrict__ out)
{
  const int nh = blockIdx.x;
  const int n = nh / H_, h = nh % H_;
  const int g = blockIdx.y;
  const int tid = threadIdx.x;
  const int lane = tid & 31, pt = tid >> 5;          // wave = pixel tile
  const int lx = lane & 15, lh = lane >> 4;
  const int WPITCH = TAPS * TAPS * 64;

  v8f acc[4] = {};
#pragma unroll
  for (int ks = 0; ks < 2; ++ks) {
    const int kc = ks * 32;
#pragma unroll
    for (int ky = 0; ky < TAPS; ++ky)
#pragma unroll
      for (int kx = 0; kx < TAPS; ++kx) {
        // B fragment: 16 contiguous K at pixel (h+ky, px+kx) in NHWC plane
        const int hh = h + ky + (TAPS == 1 ? 1 : 0);
        const int ww = pt * 16 + lx + kx + COLOFS;
        const _Float16* bp =
            s_pad + (((size_t)n * HP + hh) * WPP + ww) * SQP + kc + lh * 16;
        const v16h b = cat16(*(const v8h*)bp, *(const v8h*)(bp + 8));
#pragma unroll
        for (int ct = 0; ct < 4; ++ct) {
          const _Float16* wrow = Wq + (size_t)(g * 64 + ct * 16 + lx) * WPITCH
                                 + (ky * TAPS + kx) * 64 + kc + lh * 16;
          const v16h a = cat16(*(const v8h*)wrow, *(const v8h*)(wrow + 8));
          acc[ct] = __builtin_amdgcn_wmma_f32_16x16x32_f16(
              false, a, false, b, (short)0, acc[ct], false, false);
        }
      }
  }
  // --- epilogue: bias + ReLU + fp32 store into channel-concat NCHW output ---
  const int px = pt * 16 + lx;
  if (px < W_) {
#pragma unroll
    for (int ct = 0; ct < 4; ++ct) {
#pragma unroll
      for (int r = 0; r < 8; ++r) {
        const int oc = g * 64 + ct * 16 + lh * 8 + r;
        float v = acc[ct][r] + bias[oc];
        v = v > 0.f ? v : 0.f;
        out[((size_t)(n * 384 + CBASE + oc) * H_ + h) * W_ + px] = v;
      }
    }
  }
}

// ---------------------------------------------------------------------------
extern "C" void kernel_launch(void* const* d_in, const int* in_sizes, int n_in,
                              void* d_out, int out_size, void* d_ws, size_t ws_size,
                              hipStream_t stream) {
  const float* x    = (const float*)d_in[0];
  const float* w_sq = (const float*)d_in[1];
  const float* b_sq = (const float*)d_in[2];
  const float* w_e1 = (const float*)d_in[3];
  const float* b_e1 = (const float*)d_in[4];
  const float* w_e3 = (const float*)d_in[5];
  const float* b_e3 = (const float*)d_in[6];
  float* out = (float*)d_out;

  char* ws = (char*)d_ws;
  _Float16* q_sq  = (_Float16*)(ws);                 //  64*256*2   = 32768 B
  _Float16* q_e1  = (_Float16*)(ws + 32768);         // 192*64*2    = 24576 B
  _Float16* q_e3  = (_Float16*)(ws + 57344);         // 192*576*2   = 221184 B
  _Float16* s_pad = (_Float16*)(ws + 278528);        // NHWC padded, ~30 MB

  const size_t s_pad_bytes = (size_t)N_ * PLANEREC * SQP * sizeof(_Float16);
  hipMemsetAsync(s_pad, 0, s_pad_bytes, stream);     // zero halo (capture-safe)

  prep_kernel<<<3, 256, 0, stream>>>(w_sq, w_e1, w_e3, q_sq, q_e1, q_e3);
  squeeze_kernel<<<N_ * H_, 128, 0, stream>>>(x, b_sq, q_sq, s_pad);
  expand_kernel<1, 1, 0><<<dim3(N_ * H_, 3), 128, 0, stream>>>(s_pad, b_e1, q_e1, out);
  expand_kernel<3, 0, 192><<<dim3(N_ * H_, 3), 128, 0, stream>>>(s_pad, b_e3, q_e3, out);
}
